// ScaledAttentionHead_84817014161533
// MI455X (gfx1250) — matrix operand order repaired and checked by reading, compiled, not run on hardware
//
#include <hip/hip_runtime.h>
#include <hip/hip_bf16.h>

typedef __attribute__((ext_vector_type(16))) _Float16 v16h;
typedef __attribute__((ext_vector_type(8)))  float    v8f;

static __device__ __forceinline__ v8f wmma16x16x32(v16h a, v16h b, v8f c) {
  // (neg_a, A, neg_b, B, c_mod, C, reuse_a, reuse_b)
  return __builtin_amdgcn_wmma_f32_16x16x32_f16(false, a, false, b, (short)0, c, false, false);
}

// A-fragment (16x32 f16). Lane r=lane&15 holds matrix row r; grp=lane>>4 selects K sub-block.
// Layout: grp0 K={0..15}, grp1 K={16..31} (matches B fragment); VGPR j holds f16 pair (kb, kb+1).
static __device__ __forceinline__ v16h load_a_frag(const _Float16* rowBase, int grp, int kk) {
  union { unsigned u[8]; v16h v; } f;
  const int kOff = grp * 16 + kk;
#pragma unroll
  for (int j = 0; j < 8; ++j) {
    int kb = (j << 1) + kOff;
    f.u[j] = *(const unsigned*)(rowBase + kb);
  }
  return f.v;
}

// B-fragment (32x16 f16). Lane r holds column r; grp selects contiguous K half (16 values).
static __device__ __forceinline__ v16h load_b_frag(const _Float16* colBase, int grp, int kk) {
  union { unsigned u[8]; v16h v; } f;
  const int kOff = grp * 16 + kk;
#pragma unroll
  for (int j = 0; j < 8; ++j)
    f.u[j] = *(const unsigned*)(colBase + kOff + (j << 1));
  return f.v;
}

// XOR lane exchange within 32 lanes via ds_swizzle (group-of-32: offset = xor<<10 | and 0x1f).
static __device__ __forceinline__ float swz_xor1(float x) {
  return __int_as_float(__builtin_amdgcn_ds_swizzle(__float_as_int(x), 0x041f));
}
static __device__ __forceinline__ float swz_xor2(float x) {
  return __int_as_float(__builtin_amdgcn_ds_swizzle(__float_as_int(x), 0x081f));
}
static __device__ __forceinline__ float swz_xor4(float x) {
  return __int_as_float(__builtin_amdgcn_ds_swizzle(__float_as_int(x), 0x101f));
}
static __device__ __forceinline__ float swz_xor8(float x) {
  return __int_as_float(__builtin_amdgcn_ds_swizzle(__float_as_int(x), 0x201f));
}
static __device__ __forceinline__ float red_max16(float x) {  // max over 16-lane group
  x = fmaxf(x, swz_xor1(x)); x = fmaxf(x, swz_xor2(x));
  x = fmaxf(x, swz_xor4(x)); x = fmaxf(x, swz_xor8(x));
  return x;
}
static __device__ __forceinline__ float red_add16(float x) {  // sum over 16-lane group
  x += swz_xor1(x); x += swz_xor2(x); x += swz_xor4(x); x += swz_xor8(x);
  return x;
}

// Async global->LDS copy of 16 bytes (GLOBAL_LOAD_ASYNC_TO_LDS_B128, GV mode; ASYNCcnt).
static __device__ __forceinline__ void async_copy_b128(unsigned lds_byte_off, const void* gaddr) {
  asm volatile("global_load_async_to_lds_b128 %0, %1, off"
               :: "v"(lds_byte_off), "v"(gaddr) : "memory");
}
static __device__ __forceinline__ void wait_asynccnt0() {
  asm volatile("s_wait_asynccnt 0x0" ::: "memory");
}

// out[m, nc] = sum_k A[m,k] * W[nc,k] + bias[nc]; 128x64 block tile, 8 waves.
template <typename TA, typename TOut>
__global__ __launch_bounds__(256) void gemm_bias_wmma(
    const TA* __restrict__ A, const float* __restrict__ W,
    const float* __restrict__ bias, TOut* __restrict__ out,
    int M, int K, int NC) {
  constexpr int PAD = 36;  // 32 K + 4 pad halfs per row
  __shared__ __align__(16) _Float16 sA[128 * PAD];
  __shared__ __align__(16) _Float16 sB[64 * PAD];
  const int tid = threadIdx.x;
  const int wave = tid >> 5, lane = tid & 31, r = lane & 15, grp = lane >> 4;
  const int row0 = blockIdx.y * 128, col0 = blockIdx.x * 64;
  const int wm = wave * 16;
  v8f acc[4] = {};
  for (int kb = 0; kb < K; kb += 32) {
#pragma unroll
    for (int i = 0; i < 16; ++i) {  // A tile: 128x32, coalesced over K
      int e = tid + i * 256;
      int rr = e >> 5, cc = e & 31;
      sA[rr * PAD + cc] = (_Float16)(float)A[(size_t)(row0 + rr) * K + kb + cc];
    }
#pragma unroll
    for (int i = 0; i < 8; ++i) {   // B tile: 64x32
      int e = tid + i * 256;
      int rr = e >> 5, cc = e & 31;
      sB[rr * PAD + cc] = (_Float16)W[(size_t)(col0 + rr) * K + kb + cc];
    }
    __syncthreads();
    v16h a = load_a_frag(sA + (wm + r) * PAD, grp, 0);
#pragma unroll
    for (int t = 0; t < 4; ++t) {
      v16h b = load_b_frag(sB + (t * 16 + r) * PAD, grp, 0);
      acc[t] = wmma16x16x32(a, b, acc[t]);
    }
    __syncthreads();
  }
#pragma unroll
  for (int t = 0; t < 4; ++t)
#pragma unroll
    for (int j = 0; j < 8; ++j) {
      const int m = row0 + wm + grp * 8 + j;  // C/D layout: M = vgpr (+8 for hi lanes)
      const int c = col0 + t * 16 + r;        // N = lane&15
      out[(size_t)m * NC + c] = (TOut)(acc[t][j] + bias[c]);
    }
}

// Flash-style cosine attention. Grid: (L/128, H, N); 8 waves; each wave owns 16 query rows.
__global__ __launch_bounds__(256) void attn_kernel(
    const _Float16* __restrict__ qkv, const float* __restrict__ logit_scale,
    const float* __restrict__ learned_scale, const float* __restrict__ mask,
    _Float16* __restrict__ attn_out) {
  constexpr int L = 2048, C3 = 2304;
  constexpr int MPAD = 68;  // f32 mask tile row stride (16B-aligned rows, conflict-free reads)
  const int qb = blockIdx.x, h = blockIdx.y, n = blockIdx.z;
  const int tid = threadIdx.x;
  const int wave = tid >> 5, lane = tid & 31, r = lane & 15, grp = lane >> 4;
  const int q0 = qb * 128;
  __shared__ __align__(16) _Float16 Qs[128 * 64];  // q rows x d (L2-normalized)
  __shared__ __align__(16) _Float16 Ks[64 * 64];   // key rows x d (L2-normalized)
  __shared__ __align__(16) _Float16 Vts[64 * 72];  // d x keys (transposed, padded)
  __shared__ __align__(16) _Float16 Ps[8][16 * 64];
  __shared__ __align__(16) float    Ms[128 * MPAD];  // mask tile

  const size_t base = (size_t)n * L * C3 + (size_t)h * 64;

  {  // stage + normalize Q: 2 threads per row, 32 elems each
    const int row = tid >> 1, hf = tid & 1;
    const _Float16* src = qkv + base + (size_t)(q0 + row) * C3 + hf * 32;
    float v[32], ss = 0.f;
#pragma unroll
    for (int i = 0; i < 32; ++i) { v[i] = (float)src[i]; ss += v[i] * v[i]; }
    ss += swz_xor1(ss);
    const float inv = 1.0f / fmaxf(sqrtf(ss), 1e-12f);
#pragma unroll
    for (int i = 0; i < 32; ++i) Qs[row * 64 + hf * 32 + i] = (_Float16)(v[i] * inv);
  }
  __syncthreads();

  const _Float16* qrow = Qs + (wave * 16 + r) * 64;
  const v16h aq0 = load_a_frag(qrow, grp, 0);
  const v16h aq1 = load_a_frag(qrow, grp, 32);

  const float lsc = __expf(fminf(logit_scale[h], 4.605170185988092f));

  float m_[8], l_[8];
  v8f o[4] = {};
#pragma unroll
  for (int j = 0; j < 8; ++j) { m_[j] = -1e30f; l_[j] = 0.f; }

  for (int kb0 = 0; kb0 < L; kb0 += 64) {
    __syncthreads();
    {  // async-stage 128x64 mask tile directly into LDS (overlaps with K/V staging below)
#pragma unroll
      for (int i = 0; i < 8; ++i) {
        int f = tid + i * 256;          // 2048 x 16B chunks
        int row = f >> 4, c4 = f & 15;  // 16 chunks per row
        const void* g = (const void*)(mask + (size_t)(q0 + row) * L + kb0 + c4 * 4);
        unsigned lds_off = (unsigned)(size_t)(const void*)(Ms + row * MPAD + c4 * 4);
        async_copy_b128(lds_off, g);
      }
    }
    {  // stage + normalize K; stage V transposed: 4 threads per row, 16 elems each
      const int row = tid >> 2, qt = tid & 3;
      const _Float16* sk = qkv + base + (size_t)(kb0 + row) * C3 + 768 + qt * 16;
      float v[16], ss = 0.f;
#pragma unroll
      for (int i = 0; i < 16; ++i) { v[i] = (float)sk[i]; ss += v[i] * v[i]; }
      ss += swz_xor1(ss);
      ss += swz_xor2(ss);
      const float inv = 1.0f / fmaxf(sqrtf(ss), 1e-12f);
#pragma unroll
      for (int i = 0; i < 16; ++i) Ks[row * 64 + qt * 16 + i] = (_Float16)(v[i] * inv);
      const _Float16* sv = qkv + base + (size_t)(kb0 + row) * C3 + 1536 + qt * 16;
#pragma unroll
      for (int i = 0; i < 16; ++i) Vts[(qt * 16 + i) * 72 + row] = sv[i];
    }
    wait_asynccnt0();  // our async LDS writes landed; barrier publishes everyone's
    __syncthreads();

    // S = Qn @ Kn^T : 16x64 per wave (4 tiles, K=64 over d in two 32-steps)
    v8f s[4];
#pragma unroll
    for (int t = 0; t < 4; ++t) {
      const _Float16* krow = Ks + (t * 16 + r) * 64;
      v8f acc = {};
      acc = wmma16x16x32(aq0, load_b_frag(krow, grp, 0), acc);
      acc = wmma16x16x32(aq1, load_b_frag(krow, grp, 32), acc);
      s[t] = acc;
    }
    // logit scale + additive mask (LDS tile) + row max
    float rmax[8];
#pragma unroll
    for (int j = 0; j < 8; ++j) rmax[j] = -1e30f;
#pragma unroll
    for (int t = 0; t < 4; ++t) {
#pragma unroll
      for (int j = 0; j < 8; ++j) {
        float sv = s[t][j] * lsc + Ms[(wave * 16 + grp * 8 + j) * MPAD + t * 16 + r];
        s[t][j] = sv;
        rmax[j] = fmaxf(rmax[j], sv);
      }
    }
#pragma unroll
    for (int j = 0; j < 8; ++j) rmax[j] = red_max16(rmax[j]);  // row in 16-lane group
    float alpha[8], rs[8];
#pragma unroll
    for (int j = 0; j < 8; ++j) {
      float mn = fmaxf(m_[j], rmax[j]);
      alpha[j] = __expf(m_[j] - mn);
      m_[j] = mn;
      rs[j] = 0.f;
    }
    _Float16* Pw = Ps[wave];
#pragma unroll
    for (int t = 0; t < 4; ++t)
#pragma unroll
      for (int j = 0; j < 8; ++j) {
        float p = __expf(s[t][j] - m_[j]);
        rs[j] += p;
        Pw[(grp * 8 + j) * 64 + t * 16 + r] = (_Float16)p;  // D-layout -> row-major stage
      }
#pragma unroll
    for (int j = 0; j < 8; ++j) l_[j] = l_[j] * alpha[j] + red_add16(rs[j]);
#pragma unroll
    for (int t = 0; t < 4; ++t)
#pragma unroll
      for (int j = 0; j < 8; ++j) o[t][j] *= alpha[j];

    __syncthreads();  // make cross-lane P writes visible before fragment reads

    const v16h pa0 = load_a_frag(Pw + r * 64, grp, 0);
    const v16h pa1 = load_a_frag(Pw + r * 64, grp, 32);
#pragma unroll
    for (int t = 0; t < 4; ++t) {  // O += P @ V : K=64 over keys
      const _Float16* vtr = Vts + (t * 16 + r) * 72;
      o[t] = wmma16x16x32(pa0, load_b_frag(vtr, grp, 0), o[t]);
      o[t] = wmma16x16x32(pa1, load_b_frag(vtr, grp, 32), o[t]);
    }
  }

  const float osc = learned_scale[h];
#pragma unroll
  for (int t = 0; t < 4; ++t)
#pragma unroll
    for (int j = 0; j < 8; ++j) {
      const int row = q0 + wave * 16 + grp * 8 + j;
      const int d = t * 16 + r;
      attn_out[((size_t)n * L + row) * 768 + h * 64 + d] =
          (_Float16)(o[t][j] * (osc / l_[j]));
    }
}

extern "C" void kernel_launch(void* const* d_in, const int* in_sizes, int n_in,
                              void* d_out, int out_size, void* d_ws, size_t ws_size,
                              hipStream_t stream) {
  (void)in_sizes; (void)n_in; (void)out_size; (void)ws_size;
  const float* x             = (const float*)d_in[0];
  const float* in_proj_w     = (const float*)d_in[1];
  const float* in_proj_b     = (const float*)d_in[2];
  const float* logit_scale   = (const float*)d_in[3];
  const float* learned_scale = (const float*)d_in[4];
  const float* out_proj_w    = (const float*)d_in[5];
  const float* out_proj_b    = (const float*)d_in[6];
  const float* attn_mask     = (const float*)d_in[7];
  float* out = (float*)d_out;

  // Workspace: qkv f16 [4096 x 2304] then attn f16 [4096 x 768]  (25.2 MB total)
  _Float16* qkv  = (_Float16*)d_ws;
  _Float16* attn = qkv + (size_t)4096 * 2304;

  // 1) QKV projection: [4096,768] x [2304,768]^T -> f16
  gemm_bias_wmma<float, _Float16><<<dim3(36, 32), 256, 0, stream>>>(
      x, in_proj_w, in_proj_b, qkv, 4096, 768, 2304);
  // 2) Fused cosine attention (normalize, QK^T, softmax, PV, learned scale)
  attn_kernel<<<dim3(16, 12, 2), 256, 0, stream>>>(
      qkv, logit_scale, learned_scale, attn_mask, attn);
  // 3) Output projection: [4096,768] x [768,768]^T -> f32 d_out
  gemm_bias_wmma<_Float16, float><<<dim3(12, 32), 256, 0, stream>>>(
      attn, out_proj_w, out_proj_b, out, 4096, 768, 768);
}
